// AFlowNet_79980880986331
// MI455X (gfx1250) — compile-verified, matrix operand order checked by reading, not run on hardware
//
#include <hip/hip_runtime.h>
#include <hip/hip_bf16.h>

typedef _Float16 v16h __attribute__((ext_vector_type(16)));
typedef _Float16 v8h  __attribute__((ext_vector_type(8)));
typedef float    v8f  __attribute__((ext_vector_type(8)));

#define NEG_SLOPE 0.1f

__device__ __forceinline__ float leakyf(float v) { return v >= 0.f ? v : NEG_SLOPE * v; }

// ---------------- weight repack: OIHW f32 -> [Cout][tap][Cpad] f16 (zero-padded channels) ----------------
__global__ void repack_w_kernel(const float* __restrict__ src, _Float16* __restrict__ dst,
                                int Cin, int Cpad, int Cout) {
    int i = blockIdx.x * blockDim.x + threadIdx.x;
    int total = Cout * 9 * Cpad;
    if (i >= total) return;
    int ci = i % Cpad;
    int r  = (i / Cpad) % 9;
    int co = i / (Cpad * 9);
    float v = (ci < Cin) ? src[((size_t)co * Cin + ci) * 9 + r] : 0.f;
    dst[i] = (_Float16)v;
}

// ---------------- pack f32 NCHW -> f16 NHWC slice (channel offset into wider tensor) ----------------
__global__ void pack_f16_nhwc_kernel(const float* __restrict__ src, _Float16* __restrict__ dst,
                                     int total, int C, int Ctot, int coff, int HW) {
    int i = blockIdx.x * blockDim.x + threadIdx.x;
    if (i >= total) return;
    int p = i % HW;
    int c = (i / HW) % C;
    int n = i / (HW * C);
    dst[((size_t)n * HW + p) * Ctot + coff + c] = (_Float16)src[i];
}

// ---------------- correlation (49 ch, padded to 64) + leaky, reading f16 NHWC concat ----------------
// cat layout: (N,H,W,512): ch 0..255 = a (warped feats), ch 256..511 = b (cond feats)
__global__ void corr_leaky_nhwc_kernel(const _Float16* __restrict__ cat, _Float16* __restrict__ out,
                                       int N, int H, int W) {
    int i = blockIdx.x * blockDim.x + threadIdx.x;
    int HW = H * W;
    int total = N * HW * 64;
    if (i >= total) return;
    int d = i % 64;
    int p = (i / 64) % HW;
    int n = i / (64 * HW);
    if (d >= 49) { out[i] = (_Float16)0.f; return; }
    int x = p % W, y = p / W;
    int dy = d / 7 - 3, dx = d % 7 - 3;
    int yy = y + dy, xx = x + dx;
    float s = 0.f;
    if (yy >= 0 && yy < H && xx >= 0 && xx < W) {
        const _Float16* ap = cat + ((size_t)n * HW + p) * 512;
        const _Float16* bp = cat + ((size_t)n * HW + yy * W + xx) * 512 + 256;
        for (int c = 0; c < 256; c += 8) {
            v8h av = *(const v8h*)(ap + c);
            v8h bv = *(const v8h*)(bp + c);
#pragma unroll
            for (int e = 0; e < 8; ++e) s += (float)av[e] * (float)bv[e];
        }
    }
    s *= (1.f / 256.f);
    out[i] = (_Float16)leakyf(s);
}

// ---------------- bilinear grid sample; grid (N,2,H,W) normalized coords ----------------
// f32 output: NCHW. f16 output: NHWC at channel offset coff, channel stride Cstore.
__global__ void grid_sample_kernel(const float* __restrict__ im, const float* __restrict__ fg,
                                   float* __restrict__ o32, _Float16* __restrict__ o16,
                                   int N, int C, int H, int W, int Cstore, int coff) {
    int i = blockIdx.x * blockDim.x + threadIdx.x;
    int total = N * C * H * W;
    if (i >= total) return;
    int x = i % W;
    int y = (i / W) % H;
    int c = (i / (W * H)) % C;
    int n = i / (W * H * C);
    int HW = H * W;
    float gx = fg[((size_t)n * 2 + 0) * HW + y * W + x];
    float gy = fg[((size_t)n * 2 + 1) * HW + y * W + x];
    float fx = ((gx + 1.f) * W - 1.f) * 0.5f;
    float fy = ((gy + 1.f) * H - 1.f) * 0.5f;
    float x0f = floorf(fx), y0f = floorf(fy);
    float tx = fx - x0f, ty = fy - y0f;
    int x0 = (int)x0f, y0 = (int)y0f;
    int xi0 = min(max(x0, 0), W - 1);
    int xi1 = min(max(x0 + 1, 0), W - 1);
    int yi0 = min(max(y0, 0), H - 1);
    int yi1 = min(max(y0 + 1, 0), H - 1);
    const float* ip = im + ((size_t)n * C + c) * HW;
    float Ia = ip[yi0 * W + xi0], Ib = ip[yi1 * W + xi0];
    float Ic = ip[yi0 * W + xi1], Id = ip[yi1 * W + xi1];
    float v = Ia * (1.f - tx) * (1.f - ty) + Ib * (1.f - tx) * ty
            + Ic * tx * (1.f - ty)        + Id * tx * ty;
    if (o16) o16[((size_t)n * HW + y * W + x) * Cstore + coff + c] = (_Float16)v;
    else     o32[((size_t)n * C + c) * HW + y * W + x] = v;
}

// ---------------- apply_offset -> grid; optionally compose by sampling src (2ch NCHW f32) ----------------
__global__ void compose_offset_kernel(const float* __restrict__ off, const float* __restrict__ src,
                                      float* __restrict__ out, int N, int H, int W) {
    int i = blockIdx.x * blockDim.x + threadIdx.x;
    int total = N * H * W;
    if (i >= total) return;
    int x = i % W, y = (i / W) % H, n = i / (W * H);
    int HW = H * W;
    float ox = off[((size_t)n * 2 + 0) * HW + y * W + x];
    float oy = off[((size_t)n * 2 + 1) * HW + y * W + x];
    float xg = (x + ox) / ((W - 1.f) * 0.5f) - 1.f;
    float yg = (y + oy) / ((H - 1.f) * 0.5f) - 1.f;
    float o0, o1;
    if (src == nullptr) {
        o0 = xg; o1 = yg;
    } else {
        float fx = ((xg + 1.f) * W - 1.f) * 0.5f;
        float fy = ((yg + 1.f) * H - 1.f) * 0.5f;
        float x0f = floorf(fx), y0f = floorf(fy);
        float tx = fx - x0f, ty = fy - y0f;
        int x0 = (int)x0f, y0 = (int)y0f;
        int xi0 = min(max(x0, 0), W - 1);
        int xi1 = min(max(x0 + 1, 0), W - 1);
        int yi0 = min(max(y0, 0), H - 1);
        int yi1 = min(max(y0 + 1, 0), H - 1);
        float wa = (1.f - tx) * (1.f - ty), wb = (1.f - tx) * ty;
        float wc = tx * (1.f - ty), wd = tx * ty;
        const float* s0 = src + ((size_t)n * 2 + 0) * HW;
        const float* s1 = src + ((size_t)n * 2 + 1) * HW;
        o0 = s0[yi0 * W + xi0] * wa + s0[yi1 * W + xi0] * wb + s0[yi0 * W + xi1] * wc + s0[yi1 * W + xi1] * wd;
        o1 = s1[yi0 * W + xi0] * wa + s1[yi1 * W + xi0] * wb + s1[yi0 * W + xi1] * wc + s1[yi1 * W + xi1] * wd;
    }
    out[((size_t)n * 2 + 0) * HW + y * W + x] = o0;
    out[((size_t)n * 2 + 1) * HW + y * W + x] = o1;
}

// ---------------- 2x bilinear upsample for 2-channel flow (NCHW f32) ----------------
__global__ void upsample2x_kernel(const float* __restrict__ in, float* __restrict__ out,
                                  int N, int H, int W) {
    int Ho = 2 * H, Wo = 2 * W;
    int i = blockIdx.x * blockDim.x + threadIdx.x;
    int total = N * 2 * Ho * Wo;
    if (i >= total) return;
    int x = i % Wo, y = (i / Wo) % Ho, c = (i / (Wo * Ho)) % 2, n = i / (Wo * Ho * 2);
    float sx = (x + 0.5f) * 0.5f - 0.5f;
    float sy = (y + 0.5f) * 0.5f - 0.5f;
    float x0f = floorf(sx), y0f = floorf(sy);
    float tx = sx - x0f, ty = sy - y0f;
    int x0 = (int)x0f, y0 = (int)y0f;
    int xi0 = min(max(x0, 0), W - 1);
    int xi1 = min(max(x0 + 1, 0), W - 1);
    int yi0 = min(max(y0, 0), H - 1);
    int yi1 = min(max(y0 + 1, 0), H - 1);
    const float* ip = in + ((size_t)n * 2 + c) * H * W;
    float v = ip[yi0 * W + xi0] * (1.f - tx) * (1.f - ty)
            + ip[yi1 * W + xi0] * (1.f - tx) * ty
            + ip[yi0 * W + xi1] * tx * (1.f - ty)
            + ip[yi1 * W + xi1] * tx * ty;
    out[i] = v;
}

// ---------------- 3x3 conv (pad=1), NHWC f16, implicit GEMM on v_wmma_f32_16x16x32_f16 ----------------
// Block = 4 waves sharing ONE 16-pixel tile; Cout split across waves, NCO co-tiles per wave
// (B fragment loaded once per K-chunk, reused for NCO WMMAs; identical B addresses across
// the block's waves hit WGP$). A = 2x b128 tap-major weights; B = 2x b128 NHWC activations
// with halo zeroing via per-tap flag + vector select. Epilogue: bias+leaky, one b128 store.
template <int NCO>
__global__ __launch_bounds__(128)
void conv3x3_wmma_nhwc(const _Float16* __restrict__ X,   // NHWC, channel stride Cs
                       const _Float16* __restrict__ Wt,  // [Cout][9][Cs]
                       const float* __restrict__ bias,
                       _Float16* __restrict__ out16,     // NHWC, channel stride Cout
                       float* __restrict__ out32,        // NCHW f32 (2-ch flow)
                       int Cs, int Cout, int H, int W, int leakyOut) {
    const int lane = threadIdx.x;          // 0..31 (wave32)
    const int mn   = lane & 15;
    const int kg   = lane >> 4;
    const int co0  = (blockIdx.x * 4 + threadIdx.y) * (16 * NCO);
    if (co0 >= Cout) return;               // wave-uniform exit: EXEC all-ones at WMMA
    const int npix = H * W;
    const int p0 = blockIdx.y * 16;
    const int n = blockIdx.z;
    const int Krow = 9 * Cs;

    const bool pvalid = (p0 + mn) < npix;
    const int pn = min(p0 + mn, npix - 1);
    const int py = pn / W;
    const int px = pn - py * W;

    const _Float16* Xn = X + (size_t)n * npix * Cs;
    const _Float16* Wrow[NCO];
#pragma unroll
    for (int t = 0; t < NCO; ++t)
        Wrow[t] = Wt + (size_t)min(co0 + t * 16 + mn, Cout - 1) * Krow + kg * 8;

    v8f acc[NCO];
#pragma unroll
    for (int t = 0; t < NCO; ++t) acc[t] = (v8f){};

    const v8h vz = {};

#pragma unroll
    for (int tap = 0; tap < 9; ++tap) {
        const int ky = tap / 3 - 1;
        const int kx = tap % 3 - 1;
        const int yy = py + ky, xx = px + kx;
        const bool ok = pvalid && (yy >= 0) && (yy < H) && (xx >= 0) && (xx < W);
        const int yyc = min(max(yy, 0), H - 1);
        const int xxc = min(max(xx, 0), W - 1);
        const _Float16* Bb = Xn + ((size_t)yyc * W + xxc) * Cs + kg * 8;
        for (int cb = 0; cb < Cs; cb += 32) {
            v8h blo = *(const v8h*)(Bb + cb);
            v8h bhi = *(const v8h*)(Bb + cb + 16);
            blo = ok ? blo : vz;
            bhi = ok ? bhi : vz;
            v16h b = __builtin_shufflevector(blo, bhi, 0,1,2,3,4,5,6,7,8,9,10,11,12,13,14,15);
#pragma unroll
            for (int t = 0; t < NCO; ++t) {
                v8h alo = *(const v8h*)(Wrow[t] + tap * Cs + cb);
                v8h ahi = *(const v8h*)(Wrow[t] + tap * Cs + cb + 16);
                v16h a = __builtin_shufflevector(alo, ahi, 0,1,2,3,4,5,6,7,8,9,10,11,12,13,14,15);
                acc[t] = __builtin_amdgcn_wmma_f32_16x16x32_f16(false, a, false, b, (short)0, acc[t], false, false);
            }
        }
    }

    if (leakyOut) {
#pragma unroll
        for (int t = 0; t < NCO; ++t) {
            const int cb = co0 + t * 16 + kg * 8;
            v8h o;
#pragma unroll
            for (int r = 0; r < 8; ++r) {
                float v = acc[t][r] + bias[cb + r];
                o[r] = (_Float16)leakyf(v);
            }
            if (pvalid)
                *(v8h*)(out16 + ((size_t)n * npix + p0 + mn) * Cout + cb) = o;
        }
    } else {
#pragma unroll
        for (int r = 0; r < 8; ++r) {
            int co = kg * 8 + r;      // co0 == 0 for the 2-channel flow layer
            if (co < 2 && pvalid)
                out32[((size_t)n * 2 + co) * npix + p0 + mn] = acc[0][r] + bias[co];
        }
    }
}

// =====================================================================================

extern "C" void kernel_launch(void* const* d_in, const int* in_sizes, int n_in,
                              void* d_out, int out_size, void* d_ws, size_t ws_size,
                              hipStream_t stream) {
    const int Bn = 4, FPN = 256;
    const int HS[5]  = {8, 16, 32, 64, 128};   // level i uses warps[4-i]
    const int WSZ[5] = {6, 12, 24, 48, 96};
    const int CO[4] = {128, 64, 32, 2};
    const int CI[4] = {49, 128, 64, 32};

    // ---- workspace carve ----
    char* base = (char*)d_ws;
    size_t off = 0;
    auto carve = [&](size_t bytes) -> void* {
        void* p = base + off;
        off += (bytes + 255) & ~(size_t)255;
        return p;
    };
    _Float16* WB   = (_Float16*)carve(4245120ull * sizeof(_Float16));              // packed weights
    _Float16* CAT  = (_Float16*)carve(4ull * 128 * 96 * 512 * sizeof(_Float16));   // NHWC concat / corr inputs
    _Float16* T0   = (_Float16*)carve(4ull * 128 * 96 * 128 * sizeof(_Float16));   // NHWC ping
    _Float16* T1   = (_Float16*)carve(4ull * 128 * 96 * 128 * sizeof(_Float16));   // NHWC pong
    float*    LFIN = (float*)   carve(4ull * 2 * 256 * 192 * sizeof(float));
    float*    LFMID= (float*)   carve(4ull * 2 * 128 * 96 * sizeof(float));
    float*    FA   = (float*)   carve(4ull * 2 * 128 * 96 * sizeof(float));
    float*    FC   = (float*)   carve(4ull * 2 * 128 * 96 * sizeof(float));

    // ---- repack all conv weights ----
    _Float16*    wptr[2][5][4];
    const float* bptr[2][5][4];
    size_t woff = 0;
    for (int s = 0; s < 2; ++s)
        for (int L = 0; L < 5; ++L)
            for (int j = 0; j < 4; ++j) {
                int ci = (j == 0) ? (s == 0 ? 49 : 512) : CI[j];
                int cp = (ci + 31) & ~31;
                int co = CO[j];
                int idx = (s == 0 ? 11 : 51) + L * 8 + j * 2;
                const float* wsrc = (const float*)d_in[idx];
                bptr[s][L][j] = (const float*)d_in[idx + 1];
                wptr[s][L][j] = WB + woff;
                int ne = co * 9 * cp;
                repack_w_kernel<<<(ne + 255) / 256, 256, 0, stream>>>(wsrc, WB + woff, ci, cp, co);
                woff += (size_t)ne;
            }

    auto conv = [&](const _Float16* Xin, int Cs, int Cout, _Float16* o16, float* o32,
                    const _Float16* Wt, const float* bs, int H, int W, int lk) {
        int npix = H * W;
        dim3 blk(32, 4);
        if (Cout == 128) {
            dim3 g(1, (npix + 15) / 16, Bn);
            conv3x3_wmma_nhwc<2><<<g, blk, 0, stream>>>(Xin, Wt, bs, o16, o32, Cs, Cout, H, W, lk);
        } else {
            dim3 g((Cout + 63) / 64, (npix + 15) / 16, Bn);
            conv3x3_wmma_nhwc<1><<<g, blk, 0, stream>>>(Xin, Wt, bs, o16, o32, Cs, Cout, H, W, lk);
        }
    };

    float* OUT = (float*)d_out;
    float* OUT_FLOW = OUT + (size_t)Bn * 3 * 256 * 192;

    const float* lastflow = nullptr;
    for (int i = 0; i < 5; ++i) {
        const int k = 4 - i;
        const int H = HS[i], W = WSZ[i];
        const int npix = H * W;
        const float* xw = (const float*)d_in[1 + 2 * k];
        const float* xc = (const float*)d_in[2 + 2 * k];
        int tot = Bn * FPN * npix;

        // (a) corr inputs as f16 NHWC into CAT: a -> ch[0:256], xc -> ch[256:512]
        pack_f16_nhwc_kernel<<<(tot + 255) / 256, 256, 0, stream>>>(xc, CAT, tot, FPN, 512, 256, npix);
        if (lastflow)
            grid_sample_kernel<<<(tot + 255) / 256, 256, 0, stream>>>(
                xw, lastflow, (float*)nullptr, CAT, Bn, FPN, H, W, 512, 0);
        else
            pack_f16_nhwc_kernel<<<(tot + 255) / 256, 256, 0, stream>>>(xw, CAT, tot, FPN, 512, 0, npix);

        // (b) correlation + leaky -> T0 (NHWC, stride 64, ch 49..63 zero)
        {
            int ct = Bn * npix * 64;
            corr_leaky_nhwc_kernel<<<(ct + 255) / 256, 256, 0, stream>>>(CAT, T0, Bn, H, W);
        }

        // (c) main conv stack 49(64)->128->64->32->2
        conv(T0, 64,  128, T1, nullptr, wptr[0][i][0], bptr[0][i][0], H, W, 1);
        conv(T1, 128, 64,  T0, nullptr, wptr[0][i][1], bptr[0][i][1], H, W, 1);
        conv(T0, 64,  32,  T1, nullptr, wptr[0][i][2], bptr[0][i][2], H, W, 1);
        conv(T1, 32,  2,   nullptr, FA, wptr[0][i][3], bptr[0][i][3], H, W, 0);

        // flow = apply_offset(FA); compose with last_flow (or transpose at coarsest) -> LFMID
        {
            int ct = Bn * npix;
            compose_offset_kernel<<<(ct + 255) / 256, 256, 0, stream>>>(FA, lastflow, LFMID, Bn, H, W);
        }

        // (d) warp xw by new flow -> CAT ch[0:256] (xc half still valid from (a))
        grid_sample_kernel<<<(tot + 255) / 256, 256, 0, stream>>>(
            xw, LFMID, (float*)nullptr, CAT, Bn, FPN, H, W, 512, 0);

        // refine conv stack 512->128->64->32->2
        conv(CAT, 512, 128, T0, nullptr, wptr[1][i][0], bptr[1][i][0], H, W, 1);
        conv(T0,  128, 64,  T1, nullptr, wptr[1][i][1], bptr[1][i][1], H, W, 1);
        conv(T1,  64,  32,  T0, nullptr, wptr[1][i][2], bptr[1][i][2], H, W, 1);
        conv(T0,  32,  2,   nullptr, FA, wptr[1][i][3], bptr[1][i][3], H, W, 0);

        // flow = sample(LFMID, apply_offset(FA)) -> FC
        {
            int ct = Bn * npix;
            compose_offset_kernel<<<(ct + 255) / 256, 256, 0, stream>>>(FA, LFMID, FC, Bn, H, W);
        }

        // (e) last_flow = upsample2x(FC); final level writes straight into d_out flow region
        float* lfdst = (i == 4) ? OUT_FLOW : LFIN;
        {
            int ct = Bn * 2 * (2 * H) * (2 * W);
            upsample2x_kernel<<<(ct + 255) / 256, 256, 0, stream>>>(FC, lfdst, Bn, H, W);
        }
        lastflow = lfdst;
    }

    // x_warp = grid_sample(x, last_flow) -> d_out image region (f32 NCHW)
    {
        const float* x = (const float*)d_in[0];
        int tot = Bn * 3 * 256 * 192;
        grid_sample_kernel<<<(tot + 255) / 256, 256, 0, stream>>>(
            x, lastflow, OUT, (_Float16*)nullptr, Bn, 3, 256, 192, 3, 0);
    }
}
